// StyleEncoder_71846212927806
// MI455X (gfx1250) — matrix operand order repaired and checked by reading, not compiled, not run on hardware
//
#include <hip/hip_runtime.h>
#include <hip/hip_bf16.h>

typedef __attribute__((ext_vector_type(16))) _Float16 v16h;
typedef __attribute__((ext_vector_type(8)))  _Float16 v8h;
typedef __attribute__((ext_vector_type(8)))  float    v8f;

#define NIMG 256
#define HW   256
#define CONV_LDS(CI) (256 * ((CI) + 8) * 2)

__device__ __forceinline__ int reflect16(int i) {
  return i < 0 ? -i : (i > 15 ? 30 - i : i);
}

__device__ __forceinline__ v16h cat8(v8h lo, v8h hi) {
  return __builtin_shufflevector(lo, hi, 0, 1, 2, 3, 4, 5, 6, 7,
                                 8, 9, 10, 11, 12, 13, 14, 15);
}

// --- gfx1250 async global->LDS copy (16B per lane), tracked by ASYNCcnt ---
__device__ __forceinline__ void async_ld_b128(const _Float16* g, const _Float16* lds) {
  uint32_t loff = (uint32_t)(uintptr_t)lds;   // LDS aperture: addr[31:0] = LDS offset
  asm volatile("global_load_async_to_lds_b128 %0, %1, off"
               :: "v"(loff), "v"(g) : "memory");
}
__device__ __forceinline__ void wait_async0() {
#if __has_builtin(__builtin_amdgcn_s_wait_asynccnt)
  __builtin_amdgcn_s_wait_asynccnt(0);
#else
  asm volatile("s_wait_asynccnt 0" ::: "memory");
#endif
}

// ---------------- weight repack kernels ----------------

__global__ __launch_bounds__(256)
void f32_to_f16_kernel(const float* __restrict__ s, _Float16* __restrict__ d, int n) {
  int i = blockIdx.x * 256 + threadIdx.x;
  if (i < n) d[i] = (_Float16)s[i];
}

// OIHW 3x3 -> [tap r][co][ci] f16 so per-tap GEMM A-rows are K-contiguous
__global__ __launch_bounds__(256)
void convw_kernel(const float* __restrict__ s, _Float16* __restrict__ d, int CO, int CI) {
  int i = blockIdx.x * 256 + threadIdx.x;
  if (i >= CO * CI * 9) return;
  int co  = i / (CI * 9);
  int rem = i - co * CI * 9;
  int ci  = rem / 9;
  int r   = rem - ci * 9;
  d[((size_t)r * CO + co) * CI + ci] = (_Float16)s[i];
}

// kv_w (256 x 512) -> f16 [col][k]
__global__ __launch_bounds__(256)
void kvw_tr_kernel(const float* __restrict__ s, _Float16* __restrict__ d) {
  int i = blockIdx.x * 256 + threadIdx.x;      // 131072
  int k = i & 255, col = i >> 8;
  d[col * 256 + k] = (_Float16)s[k * 512 + col];
}

// codebook gather -> x0 f16 [img][pixel f][e]
__global__ __launch_bounds__(256)
void embed_kernel(const int* __restrict__ idx, const float* __restrict__ cbk,
                  _Float16* __restrict__ out) {
  int i = blockIdx.x * 256 + threadIdx.x;      // (img*256+f)*256+e
  int e = i & 255, f = (i >> 8) & 255, img = i >> 16;
  out[i] = (_Float16)cbk[(size_t)idx[img * 256 + f] * 256 + e];
}

// ---------------- fused IN+ReLU+reflect-pad conv: 9 accumulated WMMA GEMMs ----------------
// in/out/skip : [NIMG][256 pixels][C] f16 (channel-last)   wA : [KS*KS][CO][CI] f16
template<int CI, int CO, bool NORM, int KS, bool SKIP>
__global__ __launch_bounds__(256)
void conv_wmma_kernel(const _Float16* __restrict__ in, const _Float16* __restrict__ wA,
                      const float* __restrict__ bias, const _Float16* __restrict__ skip,
                      _Float16* __restrict__ out) {
  constexpr int LDST = CI + 8;                 // pad: breaks bank aliasing for b-frags
  extern __shared__ _Float16 smem[];           // [256 pixels][LDST]
  const int img = blockIdx.x, tid = threadIdx.x;
  const int wave = tid >> 5, lane = tid & 31;
  const _Float16* src = in + (size_t)img * 256 * CI;

  // stage 1: async DMA image tile into LDS (16B chunks, no VGPR round trip)
  constexpr int CHUNKS = 256 * CI / 8;         // multiple of 256 for all CI used
  for (int cid = tid; cid < CHUNKS; cid += 256) {
    const int f  = cid / (CI / 8);
    const int cp = (cid - f * (CI / 8)) * 8;
    async_ld_b128(src + f * CI + cp, smem + f * LDST + cp);
  }
  wait_async0();
  __syncthreads();

  // stage 2: per-(img,channel) instance-norm + ReLU in place (thread = channel)
  if (NORM) {
    for (int c = tid; c < CI; c += 256) {
      float s = 0.f, ss = 0.f;
      for (int p = 0; p < 256; ++p) {
        float v = (float)smem[p * LDST + c];
        s += v; ss += v * v;
      }
      float mean = s * (1.f / 256.f);
      float var  = ss * (1.f / 256.f) - mean * mean;
      float rstd = rsqrtf(var + 1e-5f);
      for (int p = 0; p < 256; ++p) {
        float v = ((float)smem[p * LDST + c] - mean) * rstd;
        smem[p * LDST + c] = (_Float16)(v > 0.f ? v : 0.f);
      }
    }
    __syncthreads();
  }

  // stage 3: D[co, pixel] = sum_taps W_tap[co, ci] x X_shift[ci, pixel]
  const int colN = lane & 15, half = lane >> 4;
  for (int t = wave; t < CO; t += 8) {         // (CO/16) m-tiles * 16 output rows
    const int mt = t >> 4;
    const int oy = t & 15;
    v8f acc = {};
    for (int r9 = 0; r9 < KS * KS; ++r9) {
      int pix;
      if (KS == 3) {
        const int ky = r9 / 3, kx = r9 - ky * 3;
        pix = reflect16(oy + ky - 1) * 16 + reflect16(colN + kx - 1);
      } else {
        pix = oy * 16 + colN;
      }
      const _Float16* bbase = smem + pix * LDST;
      const _Float16* arow  = wA + ((size_t)r9 * CO + mt * 16 + colN) * CI;
      #pragma unroll
      for (int k0 = 0; k0 < CI; k0 += 32) {
        const v8h a0 = *(const v8h*)(arow + k0 + half * 8);        // K {0..7|8..15}
        const v8h a1 = *(const v8h*)(arow + k0 + half * 8 + 16);   // K {16..23|24..31}
        const v8h b0 = *(const v8h*)(bbase + k0 + half * 8);       // K {0..7|8..15}
        const v8h b1 = *(const v8h*)(bbase + k0 + half * 8 + 16);  // K {16..23|24..31}
        acc = __builtin_amdgcn_wmma_f32_16x16x32_f16(false, cat8(a0, a1), false,
                                                     cat8(b0, b1), (short)0, acc,
                                                     false, false);
      }
    }
    // epilogue: D rows = 8 consecutive co -> single b128 store
    const int f  = oy * 16 + colN;
    const int c0 = mt * 16 + half * 8;
    const size_t base = ((size_t)img * 256 + f) * CO + c0;
    v8h sk = {};
    if (SKIP) sk = *(const v8h*)(skip + base);
    v8h o;
    #pragma unroll
    for (int r = 0; r < 8; ++r) {
      float v = acc[r] + bias[c0 + r];
      if (SKIP) v += (float)sk[r];
      o[r] = (_Float16)v;
    }
    *(v8h*)(out + base) = o;
  }
}

// ---------------- kv projection: feat(B*2048 x 256) @ kv_w(256 x 512) ----------------
__global__ __launch_bounds__(256)
void kv_gemm_kernel(const _Float16* __restrict__ act,  // [NIMG][256][256] f16 (act6)
                    const _Float16* __restrict__ w,    // [512][256] f16 (kv_w^T)
                    float* __restrict__ kv) {          // [B][2048][512] f32
  constexpr int LDST = 264;
  extern __shared__ _Float16 smem[];                   // [256 f][264]
  const int img = blockIdx.x, tid = threadIdx.x;
  const int wave = tid >> 5, lane = tid & 31;
  const _Float16* src = act + (size_t)img * 256 * 256;
  for (int cid = tid; cid < 8192; cid += 256) {
    const int f  = cid >> 5;
    const int cp = (cid & 31) * 8;
    async_ld_b128(src + f * 256 + cp, smem + f * LDST + cp);
  }
  wait_async0();
  __syncthreads();

  const int colN = lane & 15, half = lane >> 4;
  const int bb = img >> 3, nn = img & 7;
  for (int t = wave; t < 512; t += 8) {                // 32 col-tiles x 16 pixel-tiles
    const int mt = t >> 4, nt = t & 15;
    const int f  = nt * 16 + colN;
    v8f acc = {};
    const _Float16* bbase = smem + f * LDST;
    const _Float16* arow  = w + (size_t)(mt * 16 + colN) * 256;
    #pragma unroll
    for (int k0 = 0; k0 < 256; k0 += 32) {
      const v8h a0 = *(const v8h*)(arow + k0 + half * 8);
      const v8h a1 = *(const v8h*)(arow + k0 + half * 8 + 16);
      const v8h b0 = *(const v8h*)(bbase + k0 + half * 8);
      const v8h b1 = *(const v8h*)(bbase + k0 + half * 8 + 16);
      acc = __builtin_amdgcn_wmma_f32_16x16x32_f16(false, cat8(a0, a1), false,
                                                   cat8(b0, b1), (short)0, acc,
                                                   false, false);
    }
    float* dst = kv + ((size_t)(bb * 2048 + nn * 256 + f)) * 512 + mt * 16 + half * 8;
    #pragma unroll
    for (int r = 0; r < 8; ++r) dst[r] = acc[r];
  }
}

// ---------------- global style: sim-softmax weighted sum over references ----------------
__global__ __launch_bounds__(256)
void global_style_kernel(const _Float16* __restrict__ act, const float* __restrict__ sim,
                         float* __restrict__ xg) {     // [B][256][256] (b,f,c) f32
  __shared__ float wgt[8];
  const int b = blockIdx.y;
  if (threadIdx.x == 0) {
    float m = -1e30f;
    for (int n = 0; n < 8; ++n) m = fmaxf(m, sim[b * 8 + n] * 3.f);
    float s = 0.f, e[8];
    for (int n = 0; n < 8; ++n) { e[n] = __expf(sim[b * 8 + n] * 3.f - m); s += e[n]; }
    for (int n = 0; n < 8; ++n) wgt[n] = e[n] / s;
  }
  __syncthreads();
  const int i = blockIdx.x * 256 + threadIdx.x;        // (f,c)
  const int c = i & 255, f = i >> 8;
  float acc = 0.f;
  for (int n = 0; n < 8; ++n)
    acc += wgt[n] * (float)act[(((size_t)(b * 8 + n)) * 256 + f) * 256 + c];
  xg[((size_t)b * 256 + f) * 256 + c] = acc;
}

// ---------------- q = LN_head((ids+wpe) @ q_w) ----------------
__global__ __launch_bounds__(256)
void q_proj_kernel(const float* __restrict__ ids, const float* __restrict__ wpe,
                   const float* __restrict__ qw, const float* __restrict__ lnqw,
                   const float* __restrict__ lnqb, float* __restrict__ q) {
  __shared__ float xrow[256], qraw[256];
  const int b = blockIdx.y, l = blockIdx.x, tid = threadIdx.x;
  xrow[tid] = ids[((size_t)b * 45 + l) * 256 + tid] + wpe[l * 256 + tid];
  __syncthreads();
  float acc = 0.f;
  for (int k = 0; k < 256; ++k) acc += xrow[k] * qw[k * 256 + tid];
  qraw[tid] = acc;
  __syncthreads();
  const int h = tid >> 5, d = tid & 31;
  float s = 0.f, ss = 0.f;
  const float* hq = qraw + h * 32;
  for (int i = 0; i < 32; ++i) { s += hq[i]; ss += hq[i] * hq[i]; }
  float mean = s * (1.f / 32.f);
  float var  = ss * (1.f / 32.f) - mean * mean;
  float rstd = rsqrtf(var + 1e-5f);
  q[((size_t)b * 45 + l) * 256 + tid] = (acc - mean) * rstd * lnqw[d] + lnqb[d];
}

// ---------------- cross attention (scores in LDS) + c_proj, per (b,l) ----------------
__global__ __launch_bounds__(256)
void attn_kernel(const float* __restrict__ q, const float* __restrict__ kv,
                 const float* __restrict__ lnkw, const float* __restrict__ lnkb,
                 const float* __restrict__ cproj, float* __restrict__ xl) {
  extern __shared__ float dynf[];
  float* sc = dynf;              // [8][2048]
  float* qs = dynf + 8 * 2048;   // [256]
  float* os = qs + 256;          // [256]
  const int b = blockIdx.y, l = blockIdx.x, tid = threadIdx.x;
  qs[tid] = q[((size_t)b * 45 + l) * 256 + tid];
  __syncthreads();
  const float scale = 0.1767766952966369f;   // 1/sqrt(32)
  for (int s = tid; s < 2048; s += 256) {
    const float* krow = kv + ((size_t)b * 2048 + s) * 512;
    for (int h = 0; h < 8; ++h) {
      const float* kr = krow + h * 32;
      float m = 0.f;
      for (int d = 0; d < 32; ++d) m += kr[d];
      m *= (1.f / 32.f);
      float v = 0.f;
      for (int d = 0; d < 32; ++d) { float t = kr[d] - m; v += t * t; }
      float rstd = rsqrtf(v * (1.f / 32.f) + 1e-5f);
      float dot = 0.f;
      const float* qh = qs + h * 32;
      for (int d = 0; d < 32; ++d)
        dot += qh[d] * ((kr[d] - m) * rstd * lnkw[d] + lnkb[d]);
      sc[h * 2048 + s] = dot * scale;
    }
  }
  __syncthreads();
  {  // softmax over s, one wave32 per head
    const int h = tid >> 5, ln = tid & 31;
    float m = -1e30f;
    for (int s = ln; s < 2048; s += 32) m = fmaxf(m, sc[h * 2048 + s]);
    #pragma unroll
    for (int off = 16; off; off >>= 1) m = fmaxf(m, __shfl_xor(m, off, 32));
    float sum = 0.f;
    for (int s = ln; s < 2048; s += 32) {
      float e = __expf(sc[h * 2048 + s] - m);
      sc[h * 2048 + s] = e; sum += e;
    }
    #pragma unroll
    for (int off = 16; off; off >>= 1) sum += __shfl_xor(sum, off, 32);
    float inv = 1.f / sum;
    for (int s = ln; s < 2048; s += 32) sc[h * 2048 + s] *= inv;
  }
  __syncthreads();
  {  // o = attn @ v ; thread -> (h,d)
    const int h = tid >> 5, d = tid & 31;
    float acc = 0.f;
    const float* vcol = kv + (size_t)b * 2048 * 512 + 256 + h * 32 + d;
    for (int s = 0; s < 2048; ++s) acc += sc[h * 2048 + s] * vcol[(size_t)s * 512];
    os[tid] = acc;
  }
  __syncthreads();
  float acc = 0.f;
  for (int c = 0; c < 256; ++c) acc += os[c] * cproj[c * 256 + tid];
  xl[((size_t)b * 45 + l) * 256 + tid] = acc;
}

// ---------------- final LayerNorm (eps 1e-6, no bias) ----------------
__global__ __launch_bounds__(256)
void final_ln_kernel(const float* __restrict__ xl, const float* __restrict__ xg,
                     const float* __restrict__ lnw, float* __restrict__ out) {
  const int b = blockIdx.y, r = blockIdx.x, tid = threadIdx.x;   // r < 301
  const float* src = (r < 45) ? xl + ((size_t)b * 45 + r) * 256
                              : xg + ((size_t)b * 256 + (r - 45)) * 256;
  float v = src[tid];
  float s = v, ss = v * v;
  #pragma unroll
  for (int off = 16; off; off >>= 1) {
    s  += __shfl_xor(s,  off, 32);
    ss += __shfl_xor(ss, off, 32);
  }
  __shared__ float rs[8], rss[8];
  if ((tid & 31) == 0) { rs[tid >> 5] = s; rss[tid >> 5] = ss; }
  __syncthreads();
  if (tid == 0) {
    float a = 0.f, qv = 0.f;
    for (int i = 0; i < 8; ++i) { a += rs[i]; qv += rss[i]; }
    rs[0] = a; rss[0] = qv;
  }
  __syncthreads();
  float mean = rs[0] * (1.f / 256.f);
  float var  = rss[0] * (1.f / 256.f) - mean * mean;
  float rstd = rsqrtf(var + 1e-6f);
  out[(((size_t)b * 301) + r) * 256 + tid] = (v - mean) * rstd * lnw[tid];
}

// ---------------- host orchestration ----------------
extern "C" void kernel_launch(void* const* d_in, const int* in_sizes, int n_in,
                              void* d_out, int out_size, void* d_ws, size_t ws_size,
                              hipStream_t stream) {
  (void)in_sizes; (void)n_in; (void)out_size; (void)ws_size;
  const int*   indices  = (const int*)  d_in[0];
  const float* ids      = (const float*)d_in[1];
  const float* sim      = (const float*)d_in[2];
  const float* codebook = (const float*)d_in[3];
  const float* wpe      = (const float*)d_in[4];
  const float* qw       = (const float*)d_in[29];
  const float* kvw      = (const float*)d_in[30];
  const float* cproj    = (const float*)d_in[31];
  const float* lnqw     = (const float*)d_in[32];
  const float* lnqb     = (const float*)d_in[33];
  const float* lnkw     = (const float*)d_in[34];
  const float* lnkb     = (const float*)d_in[35];
  const float* lnw      = (const float*)d_in[36];

  _Float16* W = (_Float16*)d_ws;
  enum : size_t {
    O_CB0 = 0,       O_CB1 = 73728,   O_CB2 = 92160,
    O_R3A = 165888,  O_R3B = 313344,  O_R4A = 460800,  O_R4B = 608256,
    O_R5A = 755712,  O_R5B = 1050624, O_R5S = 1640448,
    O_R6A = 1673216, O_R6B = 2263040, O_KV  = 2852864, O_END = 2983936
  };
  // 3x3 convs -> [tap][co][ci] f16
  const struct { int di; size_t off; int co, ci; } cw[] = {
    {5,  O_CB0,  32, 256}, {7,  O_CB1,  64,  32}, {9, O_CB2, 128, 64},
    {11, O_R3A, 128, 128}, {13, O_R3B, 128, 128},
    {15, O_R4A, 128, 128}, {17, O_R4B, 128, 128},
    {19, O_R5A, 256, 128}, {21, O_R5B, 256, 256},
    {25, O_R6A, 256, 256}, {27, O_R6B, 256, 256},
  };
  for (int i = 0; i < 11; ++i) {
    int n = cw[i].co * cw[i].ci * 9;
    convw_kernel<<<(n + 255) / 256, 256, 0, stream>>>(
        (const float*)d_in[cw[i].di], W + cw[i].off, cw[i].co, cw[i].ci);
  }
  f32_to_f16_kernel<<<32768 / 256, 256, 0, stream>>>((const float*)d_in[23], W + O_R5S, 32768);
  kvw_tr_kernel<<<131072 / 256, 256, 0, stream>>>(kvw, W + O_KV);

  // f16 activations (channel-last), then f32 buffers
  _Float16* actB = W + O_END;
  _Float16* actA = actB + (size_t)16777216;
  _Float16* actC = actA + (size_t)16777216;
  float* F  = (float*)(actC + (size_t)16777216);
  float* kv = F;
  float* xg = kv + (size_t)33554432;
  float* xl = xg + 2097152;
  float* qb = xl + 368640;

  embed_kernel<<<65536, 256, 0, stream>>>(indices, codebook, actA);

  #define BIAS(i) ((const float*)d_in[i])
  conv_wmma_kernel<256, 32, true, 3, false><<<NIMG, 256, CONV_LDS(256), stream>>>(actA, W+O_CB0, BIAS(6),  nullptr, actB);
  conv_wmma_kernel< 32, 64, true, 3, false><<<NIMG, 256, CONV_LDS( 32), stream>>>(actB, W+O_CB1, BIAS(8),  nullptr, actA);
  conv_wmma_kernel< 64,128, true, 3, false><<<NIMG, 256, CONV_LDS( 64), stream>>>(actA, W+O_CB2, BIAS(10), nullptr, actB);
  // r3: x = x + cb(cb(x))
  conv_wmma_kernel<128,128, true, 3, false><<<NIMG, 256, CONV_LDS(128), stream>>>(actB, W+O_R3A, BIAS(12), nullptr, actA);
  conv_wmma_kernel<128,128, true, 3, true ><<<NIMG, 256, CONV_LDS(128), stream>>>(actA, W+O_R3B, BIAS(14), actB,   actC);
  // r4
  conv_wmma_kernel<128,128, true, 3, false><<<NIMG, 256, CONV_LDS(128), stream>>>(actC, W+O_R4A, BIAS(16), nullptr, actA);
  conv_wmma_kernel<128,128, true, 3, true ><<<NIMG, 256, CONV_LDS(128), stream>>>(actA, W+O_R4B, BIAS(18), actC,   actB);
  // r5: x = conv1x1(x) + cb(cb(x))  (128 -> 256 channels)
  conv_wmma_kernel<128,256, false,1, false><<<NIMG, 256, CONV_LDS(128), stream>>>(actB, W+O_R5S, BIAS(24), nullptr, actC);
  conv_wmma_kernel<128,256, true, 3, false><<<NIMG, 256, CONV_LDS(128), stream>>>(actB, W+O_R5A, BIAS(20), nullptr, actA);
  conv_wmma_kernel<256,256, true, 3, true ><<<NIMG, 256, CONV_LDS(256), stream>>>(actA, W+O_R5B, BIAS(22), actC,   actC);
  // r6
  conv_wmma_kernel<256,256, true, 3, false><<<NIMG, 256, CONV_LDS(256), stream>>>(actC, W+O_R6A, BIAS(26), nullptr, actA);
  conv_wmma_kernel<256,256, true, 3, true ><<<NIMG, 256, CONV_LDS(256), stream>>>(actA, W+O_R6B, BIAS(28), actC,   actB);
  #undef BIAS
  // act6 = actB
  kv_gemm_kernel<<<NIMG, 256, CONV_LDS(256), stream>>>(actB, W + O_KV, kv);

  dim3 gsg(256, 32);
  global_style_kernel<<<gsg, 256, 0, stream>>>(actB, sim, xg);
  dim3 gbl(45, 32);
  q_proj_kernel<<<gbl, 256, 0, stream>>>(ids, wpe, qw, lnqw, lnqb, qb);
  attn_kernel<<<gbl, 256, (8 * 2048 + 512) * sizeof(float), stream>>>(qb, kv, lnkw, lnkb, cproj, xl);
  dim3 gln(301, 32);
  final_ln_kernel<<<gln, 256, 0, stream>>>(xl, xg, lnw, (float*)d_out);
}
